// FlashAttention_24592982737066
// MI455X (gfx1250) — matrix operand order repaired and checked by reading, not compiled, not run on hardware
//
#include <hip/hip_runtime.h>
#include <hip/hip_bf16.h>

// ---------------------------------------------------------------------------
// MHA forward for B=2, S=2048, HID=2048, H=16, DH=128 on gfx1250 (CDNA5).
// All matmuls routed through v_wmma_f32_16x16x32_bf16 (fp32 accumulate).
// GEMM: 128x128 workgroup tile, 32x64 per-wave tile (8 WMMA / K-stage / wave).
// ---------------------------------------------------------------------------

typedef __bf16 bf16;
typedef __attribute__((ext_vector_type(16))) __bf16   v16bf;
typedef __attribute__((ext_vector_type(8)))  float    v8f;
typedef __attribute__((ext_vector_type(4)))  unsigned u32x4;

#define B_    2
#define S_    2048
#define H_    16
#define DH_   128
#define HID_  2048
#define M_    (B_ * S_)          // 4096 rows for all projections

static constexpr float SCALE  = 0.08838834764831845f;  // 1/sqrt(128)
static constexpr float LOG2E  = 1.4426950408889634f;

// float -> bf16, round-to-nearest-even, via bit manipulation (no __bf16 math).
__device__ __forceinline__ bf16 f2bf(float f) {
  union { float f; unsigned u; } x; x.f = f;
  unsigned r = x.u + 0x7FFFu + ((x.u >> 16) & 1u);
  union { unsigned short s; bf16 b; } y; y.s = (unsigned short)(r >> 16);
  return y.b;
}

union FragU { v16bf v; u32x4 u[2]; };

// A-fragment (16x32 bf16, MxK) per ISA 7.12.2:
//   lanes 0-15 : row M=lane,    k = 0..7  and 16..23
//   lanes 16-31: row M=lane-16, k = 8..15 and 24..31
// -> two contiguous 16B loads per lane from a row-major [16][ld] tile.
__device__ __forceinline__ v16bf load_frag_a(const bf16* p, int ld) {
  const int lane = threadIdx.x & 31;
  const int row  = lane & 15;
  const int koff = (lane < 16) ? 0 : 8;
  const bf16* r = p + row * ld + koff;
  FragU f;
  f.u[0] = *(const u32x4*)(r);        // k = koff .. koff+7
  f.u[1] = *(const u32x4*)(r + 16);   // k = koff+16 .. koff+23
  return f.v;
}

// B-fragment (32x16 bf16, KxN), stored as N rows of contiguous K ("[N][K]"):
//   lanes 0-15 : col N=lane,    k = 0..7  and 16..23
//   lanes 16-31: col N=lane-16, k = 8..15 and 24..31
// -> two contiguous 16B loads per lane.
__device__ __forceinline__ v16bf load_frag_b(const bf16* p, int ld) {
  const int lane = threadIdx.x & 31;
  const int n    = lane & 15;
  const int koff = (lane < 16) ? 0 : 8;
  const bf16* r = p + n * ld + koff;
  FragU f;
  f.u[0] = *(const u32x4*)(r);
  f.u[1] = *(const u32x4*)(r + 16);
  return f.v;
}

__device__ __forceinline__ v8f wmma_bf16(v16bf a, v16bf b, v8f c) {
  return __builtin_amdgcn_wmma_f32_16x16x32_bf16(false, a, false, b,
                                                 (short)0, c, false, false);
}

// cross-lane reductions within a 16-lane half-wave (xor masks < 16)
__device__ __forceinline__ float halfmax(float v) {
  #pragma unroll
  for (int m = 1; m <= 8; m <<= 1) v = fmaxf(v, __shfl_xor(v, m, 32));
  return v;
}
__device__ __forceinline__ float halfsum(float v) {
  #pragma unroll
  for (int m = 1; m <= 8; m <<= 1) v += __shfl_xor(v, m, 32);
  return v;
}

// ---------------------------------------------------------------------------
// fp32 -> bf16 cast
// ---------------------------------------------------------------------------
__global__ void cast_f32_bf16(const float* __restrict__ in,
                              bf16* __restrict__ out, int n) {
  int i = blockIdx.x * blockDim.x + threadIdx.x;
  int stride = gridDim.x * blockDim.x;
  for (; i < n; i += stride) out[i] = f2bf(in[i]);
}

// ---------------------------------------------------------------------------
// C = A[M,K] @ W[N,K]^T  (i.e. y = x @ W.T, both operands K-major in memory).
// Workgroup: 256 threads (8 waves). 128x128 output tile, K staged 32 deep.
// Wave (rw = wid>>1, cw = wid&1) owns a 32x64 sub-tile: 2 A-frags x 4 B-frags
// -> 8 WMMA per stage at 1.5 ds_load_b128 per WMMA.
// flag==0: fp32 row-major [M,N] output (final projection -> d_out)
// flag==1: bf16 head-major [B,H,S,DH] output (Q/K/V)
// ---------------------------------------------------------------------------
__global__ __launch_bounds__(256)
void gemm_bf16(const bf16* __restrict__ A, const bf16* __restrict__ W,
               float* __restrict__ Cf, bf16* __restrict__ Cb,
               int M, int N, int K, int flag) {
  __shared__ bf16 sA[128 * 40];   // 128x32 tile, ld=40 (16B-aligned rows, pad)
  __shared__ bf16 sB[128 * 40];   // 128x32 tile

  const int tid  = threadIdx.x;
  const int wid  = tid >> 5;
  const int lane = tid & 31;
  const int m0 = blockIdx.x * 128;
  const int n0 = blockIdx.y * 128;
  const int rs = (wid >> 1) * 32;   // wave row strip (32 rows) within A tile
  const int cb = (wid & 1) * 64;    // wave col base (64 cols) within B tile

  // staging mapping: 128x32 bf16 tile, 2 threads/row, 16 elems (2x16B) each
  const int srow = tid >> 1;
  const int scol = (tid & 1) * 16;

  v8f acc[2][4] = {};

  for (int k0 = 0; k0 < K; k0 += 32) {
    { // stage A tile
      const u32x4* gp = (const u32x4*)&A[(size_t)(m0 + srow) * K + k0 + scol];
      u32x4* sp = (u32x4*)&sA[srow * 40 + scol];
      sp[0] = gp[0]; sp[1] = gp[1];
    }
    { // stage B tile (W rows)
      const u32x4* gp = (const u32x4*)&W[(size_t)(n0 + srow) * K + k0 + scol];
      u32x4* sp = (u32x4*)&sB[srow * 40 + scol];
      sp[0] = gp[0]; sp[1] = gp[1];
    }
    if (k0 + 32 < K) { // prefetch next K-slab (global_prefetch_b8, no counters)
      __builtin_prefetch(&A[(size_t)(m0 + srow) * K + k0 + 32 + scol], 0, 1);
      __builtin_prefetch(&W[(size_t)(n0 + srow) * K + k0 + 32 + scol], 0, 1);
    }
    __syncthreads();

    v16bf a0 = load_frag_a(&sA[rs * 40], 40);
    v16bf a1 = load_frag_a(&sA[(rs + 16) * 40], 40);
    #pragma unroll
    for (int j = 0; j < 4; ++j) {
      v16bf b = load_frag_b(&sB[(cb + j * 16) * 40], 40);
      acc[0][j] = wmma_bf16(a0, b, acc[0][j]);
      acc[1][j] = wmma_bf16(a1, b, acc[1][j]);
    }
    __syncthreads();
  }

  // C/D layout: VGPR r, lanes 0-15 -> row r, lanes 16-31 -> row r+8; col=lane&15
  const int g   = lane >> 4;
  const int nlo = lane & 15;
  #pragma unroll
  for (int i = 0; i < 2; ++i) {
    #pragma unroll
    for (int j = 0; j < 4; ++j) {
      #pragma unroll
      for (int r = 0; r < 8; ++r) {
        int row = m0 + rs + i * 16 + r + 8 * g;
        int col = n0 + cb + j * 16 + nlo;
        float v = acc[i][j][r];
        if (flag == 0) {
          Cf[(size_t)row * N + col] = v;
        } else {
          int b = row >> 11, s = row & (S_ - 1);      // M index -> (b, s)
          int h = col >> 7,  d = col & (DH_ - 1);     // N index -> (h, d)
          Cb[(((size_t)(b * H_ + h) * S_ + s) << 7) + d] = f2bf(v);
        }
      }
    }
  }
}

// ---------------------------------------------------------------------------
// Flash attention. One WG per (b,h,128 query rows); 8 waves x 16 rows.
// Key blocks of 32 staged in LDS (K as-is; V transposed). Online softmax.
// ---------------------------------------------------------------------------
__global__ __launch_bounds__(256)
void attn_kernel(const bf16* __restrict__ Q, const bf16* __restrict__ Kk,
                 const bf16* __restrict__ V, bf16* __restrict__ O) {
  __shared__ bf16 sK [32 * 136];      // [t][d], ld=136 (pad vs 64-bank LDS)
  __shared__ bf16 sVT[128 * 40];      // [d][t] transposed, ld=40
  __shared__ bf16 sP [8 * 16 * 40];   // per-wave P bounce tile, ld=40

  const int tid  = threadIdx.x;
  const int wid  = tid >> 5;
  const int lane = tid & 31;
  const int qtile = blockIdx.x & 15;          // S_/128 = 16 query tiles
  const int bh    = blockIdx.x >> 4;          // b*H + h
  const bf16* Qb = Q  + (size_t)bh * S_ * DH_;
  const bf16* Kb = Kk + (size_t)bh * S_ * DH_;
  const bf16* Vb = V  + (size_t)bh * S_ * DH_;
  const int q0 = qtile * 128 + wid * 16;

  // Q tile (16x128) lives in registers as 4 A-fragments (global is 16B aligned)
  v16bf qf[4];
  #pragma unroll
  for (int kk = 0; kk < 4; ++kk)
    qf[kk] = load_frag_a(Qb + (size_t)q0 * DH_ + kk * 32, DH_);

  float mrun[8], lrun[8];
  v8f acc[8];
  #pragma unroll
  for (int r = 0; r < 8; ++r) { mrun[r] = -INFINITY; lrun[r] = 0.0f; }
  #pragma unroll
  for (int dt = 0; dt < 8; ++dt) acc[dt] = v8f{};

  const int g   = lane >> 4;
  const int nlo = lane & 15;
  bf16* myP = &sP[wid * 16 * 40];

  for (int t0 = 0; t0 < S_; t0 += 32) {
    const int srow = tid >> 3;
    const int scol = (tid & 7) * 16;
    { // stage K block 32x128 (rows d-contiguous -> B-fragment friendly)
      const u32x4* gp = (const u32x4*)&Kb[(size_t)(t0 + srow) * DH_ + scol];
      u32x4* sp = (u32x4*)&sK[srow * 136 + scol];
      sp[0] = gp[0]; sp[1] = gp[1];
    }
    { // stage V block transposed: sVT[d][t]
      const bf16* gp = &Vb[(size_t)(t0 + srow) * DH_ + scol];
      #pragma unroll
      for (int i = 0; i < 16; ++i) sVT[(scol + i) * 40 + srow] = gp[i];
    }
    if (t0 + 32 < S_) { // prefetch next key/value block
      __builtin_prefetch(&Kb[(size_t)(t0 + 32 + srow) * DH_ + scol], 0, 1);
      __builtin_prefetch(&Vb[(size_t)(t0 + 32 + srow) * DH_ + scol], 0, 1);
    }
    __syncthreads();

    // scores: S[16x32] = Q[16x128] @ K[t0:t0+32]^T, two 16x16 accumulators
    v8f sc0 = {}, sc1 = {};
    #pragma unroll
    for (int kk = 0; kk < 4; ++kk) {
      v16bf b0 = load_frag_b(&sK[0 * 136  + kk * 32], 136);
      v16bf b1 = load_frag_b(&sK[16 * 136 + kk * 32], 136);
      sc0 = wmma_bf16(qf[kk], b0, sc0);
      sc1 = wmma_bf16(qf[kk], b1, sc1);
    }

    // online softmax (row index == accumulator VGPR index)
    float p0[8], p1[8], alpha[8];
    #pragma unroll
    for (int r = 0; r < 8; ++r) {
      float s0 = sc0[r] * SCALE;
      float s1 = sc1[r] * SCALE;
      float mx   = halfmax(fmaxf(s0, s1));
      float mnew = fmaxf(mrun[r], mx);
      alpha[r] = exp2f((mrun[r] - mnew) * LOG2E);   // 0 on first block
      p0[r] = exp2f((s0 - mnew) * LOG2E);
      p1[r] = exp2f((s1 - mnew) * LOG2E);
      float rsum = halfsum(p0[r] + p1[r]);
      lrun[r] = lrun[r] * alpha[r] + rsum;
      mrun[r] = mnew;
    }

    // rescale O accumulators; bounce P through per-wave LDS (C-layout -> A-layout)
    #pragma unroll
    for (int r = 0; r < 8; ++r) {
      #pragma unroll
      for (int dt = 0; dt < 8; ++dt) acc[dt][r] *= alpha[r];
      myP[(r + 8 * g) * 40 + nlo]      = f2bf(p0[r]);
      myP[(r + 8 * g) * 40 + 16 + nlo] = f2bf(p1[r]);
    }
    asm volatile("s_wait_dscnt 0" ::: "memory");   // wave-local LDS RAW fence

    v16bf pa = load_frag_a(myP, 40);
    #pragma unroll
    for (int dt = 0; dt < 8; ++dt) {
      v16bf b = load_frag_b(&sVT[dt * 16 * 40], 40);
      acc[dt] = wmma_bf16(pa, b, acc[dt]);
    }
    __syncthreads();   // protect sK/sVT before next stage
  }

  // epilogue: normalize and store bf16 to [B,S,H,DH] (== [M, HID] row-major)
  const int b = bh >> 4, h = bh & 15;
  #pragma unroll
  for (int r = 0; r < 8; ++r) {
    float inv = 1.0f / lrun[r];
    int s = q0 + r + 8 * g;
    #pragma unroll
    for (int dt = 0; dt < 8; ++dt) {
      int d = dt * 16 + nlo;
      O[((size_t)(b * S_ + s) * H_ + h) * DH_ + d] = f2bf(acc[dt][r] * inv);
    }
  }
}

// ---------------------------------------------------------------------------
// Host-side launch
// ---------------------------------------------------------------------------
extern "C" void kernel_launch(void* const* d_in, const int* in_sizes, int n_in,
                              void* d_out, int out_size, void* d_ws, size_t ws_size,
                              hipStream_t stream) {
  const float* x  = (const float*)d_in[0];
  const float* wq = (const float*)d_in[1];
  const float* wk = (const float*)d_in[2];
  const float* wv = (const float*)d_in[3];
  const float* wo = (const float*)d_in[4];

  char* ws = (char*)d_ws;
  size_t off = 0;
  const size_t xBytes = (size_t)M_ * HID_ * sizeof(bf16);      // 16 MiB
  const size_t wBytes = (size_t)HID_ * HID_ * sizeof(bf16);    // 8 MiB
  bf16* xb  = (bf16*)(ws + off); off += xBytes;
  bf16* wqb = (bf16*)(ws + off); off += wBytes;
  bf16* wkb = (bf16*)(ws + off); off += wBytes;
  bf16* wvb = (bf16*)(ws + off); off += wBytes;
  bf16* wob = (bf16*)(ws + off); off += wBytes;
  bf16* Qb  = (bf16*)(ws + off); off += xBytes;
  bf16* Kb  = (bf16*)(ws + off); off += xBytes;
  bf16* Vb  = (bf16*)(ws + off); off += xBytes;
  bf16* Ab  = (bf16*)(ws + off); off += xBytes;   // total ~112 MiB

  cast_f32_bf16<<<2048, 256, 0, stream>>>(x,  xb,  M_ * HID_);
  cast_f32_bf16<<<1024, 256, 0, stream>>>(wq, wqb, HID_ * HID_);
  cast_f32_bf16<<<1024, 256, 0, stream>>>(wk, wkb, HID_ * HID_);
  cast_f32_bf16<<<1024, 256, 0, stream>>>(wv, wvb, HID_ * HID_);
  cast_f32_bf16<<<1024, 256, 0, stream>>>(wo, wob, HID_ * HID_);

  dim3 ggrid(M_ / 128, HID_ / 128);
  gemm_bf16<<<ggrid, 256, 0, stream>>>(xb, wqb, nullptr, Qb, M_, HID_, HID_, 1);
  gemm_bf16<<<ggrid, 256, 0, stream>>>(xb, wkb, nullptr, Kb, M_, HID_, HID_, 1);
  gemm_bf16<<<ggrid, 256, 0, stream>>>(xb, wvb, nullptr, Vb, M_, HID_, HID_, 1);

  attn_kernel<<<B_ * H_ * (S_ / 128), 256, 0, stream>>>(Qb, Kb, Vb, Ab);

  gemm_bf16<<<ggrid, 256, 0, stream>>>(Ab, wob, (float*)d_out, nullptr,
                                       M_, HID_, HID_, 0);
}
